// Llama4MoE_17506286698804
// MI455X (gfx1250) — compile-verified, matrix operand order checked by reading, not run on hardware
//
#include <hip/hip_runtime.h>
#include <hip/hip_bf16.h>
#include <stdint.h>

// ---------------- problem sizes ----------------
#define T_TOK 8192
#define HDIM  2048
#define IDIM  4096
#define NEXP  8

// ---------------- GEMM tiling ----------------
#define BM 64          // block tile M
#define BN 128         // block tile N
#define BK 32          // k-step (one wmma K)
#define LDT 40         // LDS row stride (ushorts): 32 data + 16B pad
#define ROWS_PAD (T_TOK + NEXP * BM)   // 8704 worst-case permuted rows
#define MT_MAX   (ROWS_PAD / BM)       // 136 M-tiles for routed path

typedef __attribute__((ext_vector_type(16))) __bf16    v16bf;
typedef __attribute__((ext_vector_type(8)))  float     v8f;
typedef __attribute__((ext_vector_type(4)))  unsigned int u32x4;
typedef __attribute__((ext_vector_type(4)))  float     f32x4;

union Frag32B { u32x4 q[2]; v16bf v; };   // 32B per-lane wmma operand

// ---------- CDNA5 async global->LDS copy (ASYNCcnt path) ----------
#if defined(__gfx1250__) && __has_builtin(__builtin_amdgcn_global_load_async_to_lds_b128)
#define HAVE_ASYNC_LDS 1
typedef __attribute__((__vector_size__(4 * sizeof(int)))) int i32v4;
typedef __attribute__((address_space(1))) i32v4 gas_i4;   // global (prints as __device__)
typedef __attribute__((address_space(3))) i32v4 lds_i4;   // LDS    (prints as __shared__)
__device__ __forceinline__ void async_b128(void* lds, const void* g) {
  // per-lane 16B copy: LDS[vdst_addr] = MEM[vaddr]; tracked by ASYNCcnt
  __builtin_amdgcn_global_load_async_to_lds_b128(
      (gas_i4*)(unsigned long long)g,
      (lds_i4*)(unsigned int)(unsigned long long)lds,
      0, 0);
}
__device__ __forceinline__ void wait_async0() {
#if __has_builtin(__builtin_amdgcn_s_wait_asynccnt)
  __builtin_amdgcn_s_wait_asynccnt(0);
#else
  asm volatile("s_wait_asynccnt 0x0" ::: "memory");
#endif
}
#else
#define HAVE_ASYNC_LDS 0
__device__ __forceinline__ void wait_async0() {}
#endif

__device__ __forceinline__ unsigned short f2bf(float f) {
  unsigned int u = __float_as_uint(f);
  u += 0x7fffu + ((u >> 16) & 1u);        // round-to-nearest-even
  return (unsigned short)(u >> 16);
}

__device__ __forceinline__ v8f bwmma(v16bf a, v16bf b, v8f c) {
  return __builtin_amdgcn_wmma_f32_16x16x32_bf16(
      /*neg_a=*/false, a, /*neg_b=*/false, b,
      /*c_mod=*/(short)0, c, /*reuse_a=*/false, /*reuse_b=*/false);
}

// ---------------- init: zero Xe (incl. padding rows), perm=-1, counts=0 ----
__global__ __launch_bounds__(256) void init_kernel(
    u32x4* __restrict__ XeV, size_t n4,
    int* __restrict__ perm, int* __restrict__ counts, int* __restrict__ mapShared)
{
  size_t gid = (size_t)blockIdx.x * blockDim.x + threadIdx.x;
  size_t gs  = (size_t)gridDim.x * blockDim.x;
  u32x4 z = {0u, 0u, 0u, 0u};
  for (size_t i = gid; i < n4; i += gs) XeV[i] = z;
  if (gid < ROWS_PAD) perm[gid] = -1;
  if (gid < NEXP)     counts[gid] = 0;
  if (gid < T_TOK / BM) mapShared[gid] = 0;   // shared path: "expert 0", stride 0
}

// ---------------- router: wave-per-token, top-1 + sigmoid ----------------
__global__ __launch_bounds__(256) void router_kernel(
    const float* __restrict__ x, const float* __restrict__ rw,
    int* __restrict__ eid, float* __restrict__ wt, int* __restrict__ counts)
{
  const int lane = threadIdx.x & 31;
  const int t = blockIdx.x * 8 + (threadIdx.x >> 5);
  float acc[NEXP];
#pragma unroll
  for (int e = 0; e < NEXP; ++e) acc[e] = 0.f;
  const float* xr = x + (size_t)t * HDIM;
  for (int h = lane; h < HDIM; h += 32) {
    float xv = xr[h];
#pragma unroll
    for (int e = 0; e < NEXP; ++e) acc[e] += xv * rw[e * HDIM + h];
  }
#pragma unroll
  for (int e = 0; e < NEXP; ++e)
    for (int off = 16; off > 0; off >>= 1) acc[e] += __shfl_down(acc[e], off);
  if (lane == 0) {
    int be = 0; float bv = acc[0];
#pragma unroll
    for (int e = 1; e < NEXP; ++e) if (acc[e] > bv) { bv = acc[e]; be = e; }
    eid[t] = be;
    wt[t]  = 1.0f / (1.0f + __expf(-bv));
    atomicAdd(&counts[be], 1);
  }
}

// ---------------- scan: 64-aligned expert segments + tile->expert map -----
__global__ void scan_kernel(const int* __restrict__ counts,
                            int* __restrict__ segStart, int* __restrict__ cursor,
                            int* __restrict__ mapRouted)
{
  if (threadIdx.x == 0 && blockIdx.x == 0) {
    int starts[NEXP], pads[NEXP];
    int off = 0;
    for (int e = 0; e < NEXP; ++e) {
      starts[e] = off;
      int pad = ((counts[e] + BM - 1) / BM) * BM;
      pads[e] = pad;
      segStart[e] = off;
      cursor[e] = 0;
      off += pad;
    }
    for (int i = 0; i < MT_MAX; ++i) mapRouted[i] = -1;
    for (int e = 0; e < NEXP; ++e) {
      int t0 = starts[e] / BM, nt = pads[e] / BM;
      for (int j = 0; j < nt; ++j) mapRouted[t0 + j] = e;
    }
  }
}

// ---------------- scatter: build bf16 Xs and permuted, weight-scaled Xe ---
__global__ __launch_bounds__(256) void scatter_kernel(
    const float* __restrict__ x, const int* __restrict__ eid, const float* __restrict__ wt,
    const int* __restrict__ segStart, int* __restrict__ cursor,
    int* __restrict__ perm, unsigned short* __restrict__ Xe, unsigned short* __restrict__ Xs)
{
  const int lane = threadIdx.x & 31;
  const int t = blockIdx.x * 8 + (threadIdx.x >> 5);
  int pos = 0; float w = 0.f;
  if (lane == 0) {
    int e = eid[t];
    w = wt[t];
    pos = segStart[e] + atomicAdd(&cursor[e], 1);
    perm[pos] = t;
  }
  pos = __shfl(pos, 0);
  w   = __shfl(w, 0);
  const float* xr = x + (size_t)t * HDIM;
  unsigned short* xe = Xe + (size_t)pos * HDIM;
  unsigned short* xs = Xs + (size_t)t * HDIM;
  for (int h = lane; h < HDIM; h += 32) {
    float xv = xr[h];
    xe[h] = f2bf(xv * w);
    xs[h] = f2bf(xv);
  }
}

// ---------------- fp32 -> bf16 weight conversion (vectorized) -------------
__global__ __launch_bounds__(256) void cvt_kernel(
    const float* __restrict__ src, unsigned short* __restrict__ dst, size_t n8)
{
  size_t gid = (size_t)blockIdx.x * blockDim.x + threadIdx.x;
  size_t gs  = (size_t)gridDim.x * blockDim.x;
  const f32x4* s4 = (const f32x4*)src;
  u32x4* d4 = (u32x4*)dst;
  for (size_t i = gid; i < n8; i += gs) {
    f32x4 a = s4[2 * i], b = s4[2 * i + 1];
    union { unsigned short h[8]; u32x4 q; } o;
    o.h[0] = f2bf(a.x); o.h[1] = f2bf(a.y); o.h[2] = f2bf(a.z); o.h[3] = f2bf(a.w);
    o.h[4] = f2bf(b.x); o.h[5] = f2bf(b.y); o.h[6] = f2bf(b.z); o.h[7] = f2bf(b.w);
    d4[i] = o.q;
  }
}

// ---------------- fused gate/up GEMM + silu epilogue ----------------------
// C[m,n] = sum_k A[m,k] * W[n,k]  (TN, both K-contiguous), dual B (gate, up)
// Double-buffered LDS; async global->LDS staging when available.
__global__ __launch_bounds__(256) void gateup_kernel(
    const unsigned short* __restrict__ A,    // [*, K] bf16 rows
    const unsigned short* __restrict__ Wg,   // [(e,) N, K] bf16
    const unsigned short* __restrict__ Wu,
    unsigned short* __restrict__ Hout,       // [*, N] bf16
    const int* __restrict__ tileMap,
    long long eStride, int K, int N)
{
  const int mt = blockIdx.y;
  const int e  = tileMap[mt];
  if (e < 0) return;
  const unsigned short* wg = Wg + (size_t)e * (size_t)eStride;
  const unsigned short* wu = Wu + (size_t)e * (size_t)eStride;
  const int m0 = mt * BM;
  const int n0 = blockIdx.x * BN;

  __shared__ unsigned short sA[2][BM * LDT];
  __shared__ unsigned short sG[2][BN * LDT];
  __shared__ unsigned short sU[2][BN * LDT];

  const int tid  = threadIdx.x;
  const int lane = tid & 31;
  const int wm   = (tid >> 5) >> 2;     // 0..1
  const int wn   = (tid >> 5) & 3;      // 0..3
  const int lm   = lane & 15;
  const int lk   = (lane >> 4) << 4;    // 0 or 16

  const int ar = tid >> 2, ac = (tid & 3) << 3;   // A tile 64x32, 8 elems/thr
  const int br = tid >> 1, bc = (tid & 1) << 4;   // B tile 128x32, 16 elems/thr

  const unsigned short* gA = A  + (size_t)(m0 + ar) * K + ac;
  const unsigned short* gG = wg + (size_t)(n0 + br) * K + bc;
  const unsigned short* gU = wu + (size_t)(n0 + br) * K + bc;

  auto stage = [&](int buf, int kk) {
#if HAVE_ASYNC_LDS
    async_b128(&sA[buf][ar * LDT + ac],     gA + kk);
    async_b128(&sG[buf][br * LDT + bc],     gG + kk);
    async_b128(&sG[buf][br * LDT + bc + 8], gG + kk + 8);
    async_b128(&sU[buf][br * LDT + bc],     gU + kk);
    async_b128(&sU[buf][br * LDT + bc + 8], gU + kk + 8);
#else
    *(u32x4*)&sA[buf][ar * LDT + ac]     = *(const u32x4*)(gA + kk);
    *(u32x4*)&sG[buf][br * LDT + bc]     = *(const u32x4*)(gG + kk);
    *(u32x4*)&sG[buf][br * LDT + bc + 8] = *(const u32x4*)(gG + kk + 8);
    *(u32x4*)&sU[buf][br * LDT + bc]     = *(const u32x4*)(gU + kk);
    *(u32x4*)&sU[buf][br * LDT + bc + 8] = *(const u32x4*)(gU + kk + 8);
#endif
    if (kk + BK < K) {
      __builtin_prefetch(gA + kk + BK, 0, 3);
      __builtin_prefetch(gG + kk + BK, 0, 3);
      __builtin_prefetch(gU + kk + BK, 0, 3);
    }
  };

  v8f accG[2][2] = {}; v8f accU[2][2] = {};

  stage(0, 0);
  wait_async0();
  __syncthreads();

  for (int kk = 0; kk < K; kk += BK) {
    const int cur = (kk >> 5) & 1;
    const int nxt = cur ^ 1;
    if (kk + BK < K) stage(nxt, kk + BK);   // overlap next fetch with compute

    Frag32B a[2], bg[2], bu[2];
#pragma unroll
    for (int i = 0; i < 2; ++i) {
      const unsigned short* p = &sA[cur][(wm * 32 + i * 16 + lm) * LDT + lk];
      a[i].q[0] = *(const u32x4*)p; a[i].q[1] = *(const u32x4*)(p + 8);
    }
#pragma unroll
    for (int j = 0; j < 2; ++j) {
      const unsigned short* pg = &sG[cur][(wn * 32 + j * 16 + lm) * LDT + lk];
      bg[j].q[0] = *(const u32x4*)pg; bg[j].q[1] = *(const u32x4*)(pg + 8);
      const unsigned short* pu = &sU[cur][(wn * 32 + j * 16 + lm) * LDT + lk];
      bu[j].q[0] = *(const u32x4*)pu; bu[j].q[1] = *(const u32x4*)(pu + 8);
    }
#pragma unroll
    for (int i = 0; i < 2; ++i)
#pragma unroll
      for (int j = 0; j < 2; ++j) {
        accG[i][j] = bwmma(a[i].v, bg[j].v, accG[i][j]);
        accU[i][j] = bwmma(a[i].v, bu[j].v, accU[i][j]);
      }

    wait_async0();        // next tile landed in LDS
    __syncthreads();      // all waves done reading cur + see next tile
  }

  const int hi8 = (lane >> 4) << 3;   // C layout: lanes 16-31 hold M+8
#pragma unroll
  for (int i = 0; i < 2; ++i)
#pragma unroll
    for (int j = 0; j < 2; ++j) {
      const int gm = m0 + wm * 32 + i * 16 + hi8;
      const int gn = n0 + wn * 32 + j * 16 + lm;
#pragma unroll
      for (int r = 0; r < 8; ++r) {
        float g = accG[i][j][r];
        float u = accU[i][j][r];
        float s = g / (1.0f + __expf(-g));     // silu
        Hout[(size_t)(gm + r) * N + gn] = f2bf(s * u);
      }
    }
}

// ---------------- down-projection GEMM ------------------------------------
// perm == nullptr : Out[row,n] = v          (shared expert, initializes out)
// perm != nullptr : Out[perm[row],n] += v   (routed, unique rows -> race-free)
__global__ __launch_bounds__(256) void down_kernel(
    const unsigned short* __restrict__ A,    // [*, K] bf16 (H buffer)
    const unsigned short* __restrict__ Wd,   // [(e,) N, K] bf16
    float* __restrict__ Out,                 // [T, N] fp32
    const int* __restrict__ tileMap,
    const int* __restrict__ perm,
    long long eStride, int K, int N)
{
  const int mt = blockIdx.y;
  const int e  = tileMap[mt];
  if (e < 0) return;
  const unsigned short* wd = Wd + (size_t)e * (size_t)eStride;
  const int m0 = mt * BM;
  const int n0 = blockIdx.x * BN;

  __shared__ unsigned short sA[2][BM * LDT];
  __shared__ unsigned short sB[2][BN * LDT];

  const int tid  = threadIdx.x;
  const int lane = tid & 31;
  const int wm   = (tid >> 5) >> 2;
  const int wn   = (tid >> 5) & 3;
  const int lm   = lane & 15;
  const int lk   = (lane >> 4) << 4;

  const int ar = tid >> 2, ac = (tid & 3) << 3;
  const int br = tid >> 1, bc = (tid & 1) << 4;

  const unsigned short* gA = A  + (size_t)(m0 + ar) * K + ac;
  const unsigned short* gB = wd + (size_t)(n0 + br) * K + bc;

  auto stage = [&](int buf, int kk) {
#if HAVE_ASYNC_LDS
    async_b128(&sA[buf][ar * LDT + ac],     gA + kk);
    async_b128(&sB[buf][br * LDT + bc],     gB + kk);
    async_b128(&sB[buf][br * LDT + bc + 8], gB + kk + 8);
#else
    *(u32x4*)&sA[buf][ar * LDT + ac]     = *(const u32x4*)(gA + kk);
    *(u32x4*)&sB[buf][br * LDT + bc]     = *(const u32x4*)(gB + kk);
    *(u32x4*)&sB[buf][br * LDT + bc + 8] = *(const u32x4*)(gB + kk + 8);
#endif
    if (kk + BK < K) {
      __builtin_prefetch(gA + kk + BK, 0, 3);
      __builtin_prefetch(gB + kk + BK, 0, 3);
    }
  };

  v8f acc[2][2] = {};

  stage(0, 0);
  wait_async0();
  __syncthreads();

  for (int kk = 0; kk < K; kk += BK) {
    const int cur = (kk >> 5) & 1;
    const int nxt = cur ^ 1;
    if (kk + BK < K) stage(nxt, kk + BK);

    Frag32B a[2], b[2];
#pragma unroll
    for (int i = 0; i < 2; ++i) {
      const unsigned short* p = &sA[cur][(wm * 32 + i * 16 + lm) * LDT + lk];
      a[i].q[0] = *(const u32x4*)p; a[i].q[1] = *(const u32x4*)(p + 8);
    }
#pragma unroll
    for (int j = 0; j < 2; ++j) {
      const unsigned short* p = &sB[cur][(wn * 32 + j * 16 + lm) * LDT + lk];
      b[j].q[0] = *(const u32x4*)p; b[j].q[1] = *(const u32x4*)(p + 8);
    }
#pragma unroll
    for (int i = 0; i < 2; ++i)
#pragma unroll
      for (int j = 0; j < 2; ++j)
        acc[i][j] = bwmma(a[i].v, b[j].v, acc[i][j]);

    wait_async0();
    __syncthreads();
  }

  const int hi8 = (lane >> 4) << 3;
#pragma unroll
  for (int i = 0; i < 2; ++i)
#pragma unroll
    for (int j = 0; j < 2; ++j) {
      const int gm = m0 + wm * 32 + i * 16 + hi8;
      const int gn = n0 + wn * 32 + j * 16 + lm;
#pragma unroll
      for (int r = 0; r < 8; ++r) {
        float v = acc[i][j][r];
        int row = gm + r;
        if (perm) {
          int t = perm[row];
          if (t >= 0) {
            float* o = &Out[(size_t)t * N + gn];
            *o = *o + v;
          }
        } else {
          Out[(size_t)row * N + gn] = v;
        }
      }
    }
}

// ---------------- host launcher -------------------------------------------
extern "C" void kernel_launch(void* const* d_in, const int* in_sizes, int n_in,
                              void* d_out, int out_size, void* d_ws, size_t ws_size,
                              hipStream_t stream)
{
  (void)in_sizes; (void)n_in; (void)out_size; (void)ws_size;

  const float* x   = (const float*)d_in[0];   // [T, H]
  const float* rw  = (const float*)d_in[1];   // [E, H]
  const float* wg  = (const float*)d_in[2];   // [E, I, H]
  const float* wu  = (const float*)d_in[3];   // [E, I, H]
  const float* wd  = (const float*)d_in[4];   // [E, H, I]
  const float* swg = (const float*)d_in[5];   // [I, H]
  const float* swu = (const float*)d_in[6];   // [I, H]
  const float* swd = (const float*)d_in[7];   // [H, I]
  float* out = (float*)d_out;                 // [T, H] fp32

  const size_t EIH = (size_t)NEXP * IDIM * HDIM;   // 67.1M elems
  const size_t IH  = (size_t)IDIM * HDIM;          // 8.39M elems

  // Workspace carve-up (~594 MB total)
  char* ws = (char*)d_ws;
  size_t off = 0;
  auto alloc = [&](size_t bytes) -> char* {
    off = (off + 255) & ~(size_t)255;
    char* p = ws + off;
    off += bytes;
    return p;
  };
  unsigned short* wgB  = (unsigned short*)alloc(EIH * 2);
  unsigned short* wuB  = (unsigned short*)alloc(EIH * 2);
  unsigned short* wdB  = (unsigned short*)alloc(EIH * 2);
  unsigned short* swgB = (unsigned short*)alloc(IH * 2);
  unsigned short* swuB = (unsigned short*)alloc(IH * 2);
  unsigned short* swdB = (unsigned short*)alloc(IH * 2);
  unsigned short* XsB  = (unsigned short*)alloc((size_t)T_TOK * HDIM * 2);
  unsigned short* XeB  = (unsigned short*)alloc((size_t)ROWS_PAD * HDIM * 2);
  unsigned short* Hb   = (unsigned short*)alloc((size_t)ROWS_PAD * IDIM * 2);
  int*   eid      = (int*)alloc(T_TOK * 4);
  float* wt       = (float*)alloc(T_TOK * 4);
  int*   perm     = (int*)alloc(ROWS_PAD * 4);
  int*   counts   = (int*)alloc(NEXP * 4);
  int*   cursor   = (int*)alloc(NEXP * 4);
  int*   segStart = (int*)alloc(NEXP * 4);
  int*   mapR     = (int*)alloc(MT_MAX * 4);
  int*   mapS     = (int*)alloc((T_TOK / BM) * 4);

  // 1) init scratch (zero padded Xe, perm=-1, counts=0, shared map=0)
  init_kernel<<<4096, 256, 0, stream>>>((u32x4*)XeB,
      (size_t)ROWS_PAD * HDIM / 8, perm, counts, mapS);

  // 2) router: top-1 expert + sigmoid weight per token
  router_kernel<<<T_TOK / 8, 256, 0, stream>>>(x, rw, eid, wt, counts);

  // 3) expert segment offsets (64-aligned) + M-tile -> expert map
  scan_kernel<<<1, 32, 0, stream>>>(counts, segStart, cursor, mapR);

  // 4) gather tokens per expert: Xe = bf16(x * w) permuted, Xs = bf16(x)
  scatter_kernel<<<T_TOK / 8, 256, 0, stream>>>(x, eid, wt, segStart, cursor,
                                                perm, XeB, XsB);

  // 5) fp32 -> bf16 weight conversion (once per call)
  cvt_kernel<<<8192, 256, 0, stream>>>(wg,  wgB,  EIH / 8);
  cvt_kernel<<<8192, 256, 0, stream>>>(wu,  wuB,  EIH / 8);
  cvt_kernel<<<8192, 256, 0, stream>>>(wd,  wdB,  EIH / 8);
  cvt_kernel<<<2048, 256, 0, stream>>>(swg, swgB, IH / 8);
  cvt_kernel<<<2048, 256, 0, stream>>>(swu, swuB, IH / 8);
  cvt_kernel<<<2048, 256, 0, stream>>>(swd, swdB, IH / 8);

  // 6) shared expert: H = silu(Xs@swg^T) * (Xs@swu^T); out = H@swd^T (write)
  gateup_kernel<<<dim3(IDIM / BN, T_TOK / BM), 256, 0, stream>>>(
      XsB, swgB, swuB, Hb, mapS, 0, HDIM, IDIM);
  down_kernel<<<dim3(HDIM / BN, T_TOK / BM), 256, 0, stream>>>(
      Hb, swdB, out, mapS, nullptr, 0, IDIM, HDIM);

  // 7) routed experts over compacted segments; out[perm[row]] += (add)
  gateup_kernel<<<dim3(IDIM / BN, MT_MAX), 256, 0, stream>>>(
      XeB, wgB, wuB, Hb, mapR, (long long)IDIM * HDIM, HDIM, IDIM);
  down_kernel<<<dim3(HDIM / BN, MT_MAX), 256, 0, stream>>>(
      Hb, wdB, out, mapR, perm, (long long)HDIM * IDIM, IDIM, HDIM);
}
